// Encoder_11708080849202
// MI455X (gfx1250) — compile-verified, matrix operand order checked by reading, not run on hardware
//
#include <hip/hip_runtime.h>

#define NNODES 50000
#define NEDGES 800000
#define CH 128  // IN_CH == HID_CH == concat(OUT,OUT) == 128

typedef __bf16 bf16_t;
typedef bf16_t v16bf __attribute__((ext_vector_type(16)));
typedef float  v8f   __attribute__((ext_vector_type(8)));

__device__ __forceinline__ bf16_t bf16_bits(unsigned short u) {
  union { unsigned short s; bf16_t b; } c; c.s = u; return c.b;
}

// Split fp32 into bf16 hi (truncation, so x - hi is exact) + bf16 lo.
__device__ __forceinline__ void split_bf16(float x, bf16_t& hi, bf16_t& lo) {
  unsigned int u = __float_as_uint(x);
  unsigned short h = (unsigned short)(u >> 16);
  float hf = __uint_as_float(((unsigned int)h) << 16);
  unsigned short l = (unsigned short)(__float_as_uint(x - hf) >> 16);
  hi = bf16_bits(h);
  lo = bf16_bits(l);
}

// ---------------------------------------------------------------------------
// Pre-pack a 128x128 row-major f32 weight matrix into WMMA B-fragment order,
// split into bf16 hi/lo. Fragment index: ((ktile*8 + ntile)*32 + lane), each
// entry is a v16bf (32B). B-matrix 32x16 bf16 lane layout (n = lane&15,
// half = lane>>4): e=0..7 -> K = half*8+e ; e=8..15 -> K = 16+half*8+(e-8).
// 1024 threads total (4 blocks x 256).
// ---------------------------------------------------------------------------
__global__ void k_pack_wfrag(const float* __restrict__ W,
                             v16bf* __restrict__ Whi, v16bf* __restrict__ Wlo) {
  int t = blockIdx.x * 256 + threadIdx.x;
  if (t >= 4 * 8 * 32) return;
  int ktile = t >> 8;          // 0..3
  int ntile = (t >> 5) & 7;    // 0..7
  int lane  = t & 31;
  int half  = lane >> 4;
  int n     = ntile * 16 + (lane & 15);
  v16bf hv, lv;
#pragma unroll
  for (int e = 0; e < 16; ++e) {
    int koff = (e < 8) ? (half * 8 + e) : (16 + half * 8 + (e - 8));
    int k = ktile * 32 + koff;
    bf16_t h, l;
    split_bf16(W[k * CH + n], h, l);
    hv[e] = h;
    lv[e] = l;
  }
  int idx = (ktile * 8 + ntile) * 32 + lane;
  Whi[idx] = hv;
  Wlo[idx] = lv;
}

// ---------------------------------------------------------------------------
// WMMA GEMM: C[M,128] = A[M,128] * B[128,128], fp32 in/out, bf16x3 split.
// B pre-packed into fragment order (hi/lo). Block = 256 threads = 8 waves;
// block handles 16 rows, wave w handles cols [16w,16w+16). M%16==0.
// A 16-bit 16x32 lane layout: m = lane&15, half = lane>>4,
//   e=0..7 -> K = kt + half*8 + e ; e=8..15 -> K = kt + 16 + half*8 + (e-8)
// C/D f32: VGPR r, lane: row = r + 8*(lane>>4), col = lane&15.
// ---------------------------------------------------------------------------
__global__ __launch_bounds__(256)
void gcn_gemm_bf16x3(const float* __restrict__ A,
                     const v16bf* __restrict__ Bhi, const v16bf* __restrict__ Blo,
                     float* __restrict__ C, int M) {
  const int wave = threadIdx.x >> 5;
  const int lane = threadIdx.x & 31;
  const int half = lane >> 4;
  const int ml   = lane & 15;
  const int m0   = blockIdx.x << 4;

  const float* arow0 = A + (size_t)(m0 + ml) * CH + half * 8;

  v8f acc = {};
#pragma unroll
  for (int kt4 = 0; kt4 < 4; ++kt4) {
    const float* arow = arow0 + kt4 * 32;
    float4 a0 = *(const float4*)(arow);
    float4 a1 = *(const float4*)(arow + 4);
    float4 a2 = *(const float4*)(arow + 16);
    float4 a3 = *(const float4*)(arow + 20);
    float af[16] = {a0.x, a0.y, a0.z, a0.w, a1.x, a1.y, a1.z, a1.w,
                    a2.x, a2.y, a2.z, a2.w, a3.x, a3.y, a3.z, a3.w};
    v16bf ahi, alo;
#pragma unroll
    for (int e = 0; e < 16; ++e) {
      bf16_t h, l;
      split_bf16(af[e], h, l);
      ahi[e] = h;
      alo[e] = l;
    }
    int idx = (kt4 * 8 + wave) * 32 + lane;
    v16bf bhi = Bhi[idx];
    v16bf blo = Blo[idx];
    acc = __builtin_amdgcn_wmma_f32_16x16x32_bf16(false, ahi, false, bhi,
                                                  (short)0, acc, false, false);
    acc = __builtin_amdgcn_wmma_f32_16x16x32_bf16(false, alo, false, bhi,
                                                  (short)0, acc, false, false);
    acc = __builtin_amdgcn_wmma_f32_16x16x32_bf16(false, ahi, false, blo,
                                                  (short)0, acc, false, false);
  }
  float* crow = C + (size_t)m0 * CH + wave * 16;
#pragma unroll
  for (int r = 0; r < 8; ++r)
    crow[(size_t)(r + half * 8) * CH + ml] = acc[r];
}

// ---------------------------------------------------------------------------
// Degree / normalization
// ---------------------------------------------------------------------------
__global__ void k_deg_init(float* __restrict__ deg, int n) {
  int i = blockIdx.x * 256 + threadIdx.x;
  if (i < n) deg[i] = 1.0f;  // self-loop
}

__global__ void k_deg_count(const int* __restrict__ cols, float* __restrict__ deg, int e) {
  int i = blockIdx.x * 256 + threadIdx.x;
  if (i < e) atomicAdd(&deg[cols[i]], 1.0f);
}

__global__ void k_dinv(float* __restrict__ d, int n) {
  int i = blockIdx.x * 256 + threadIdx.x;
  if (i < n) d[i] = rsqrtf(d[i]);  // deg >= 1 always (self-loop)
}

// ---------------------------------------------------------------------------
// out[c][j] = dinv[c]^2 * h[c][j] + bias[j]   (self-loop term + bias; fully
// initializes the accumulation buffer). bias = [bLo(0..63) | bHi(0..63)].
// One thread per float4 => n*32 threads.
// ---------------------------------------------------------------------------
__global__ void k_self_bias(const float* __restrict__ h, const float* __restrict__ dinv,
                            const float* __restrict__ bLo, const float* __restrict__ bHi,
                            float* __restrict__ out, int n) {
  int t = blockIdx.x * 256 + threadIdx.x;
  if (t >= n * 32) return;
  int node = t >> 5;
  int j = (t & 31) << 2;
  float d = dinv[node];
  float s = d * d;
  const float4 hv = *(const float4*)(h + (size_t)node * CH + j);
  const float4 bv = *(const float4*)((j < 64) ? (bLo + j) : (bHi + (j - 64)));
  float4 o;
  o.x = s * hv.x + bv.x;
  o.y = s * hv.y + bv.y;
  o.z = s * hv.z + bv.z;
  o.w = s * hv.w + bv.w;
  *(float4*)(out + (size_t)node * CH + j) = o;
}

// ---------------------------------------------------------------------------
// Edge scatter: out[col] += dinv[row]*dinv[col] * h[row].  One wave per edge,
// lane handles 4 contiguous channels (coalesced 512B row gather per wave).
// Non-returning fp32 atomics; h/out are L2-resident (51MB << 192MB L2).
// ---------------------------------------------------------------------------
__global__ void k_edge_scatter(const int* __restrict__ rows, const int* __restrict__ cols,
                               const float* __restrict__ dinv, const float* __restrict__ h,
                               float* __restrict__ out, int e) {
  unsigned t = blockIdx.x * 256u + threadIdx.x;
  unsigned ed = t >> 5;
  if (ed >= (unsigned)e) return;
  int j = (t & 31u) << 2;
  int r = rows[ed];
  int c = cols[ed];
  float w = dinv[r] * dinv[c];
  const float4 hv = *(const float4*)(h + (size_t)r * CH + j);
  float* o = out + (size_t)c * CH + j;
  atomicAdd(o + 0, w * hv.x);
  atomicAdd(o + 1, w * hv.y);
  atomicAdd(o + 2, w * hv.z);
  atomicAdd(o + 3, w * hv.w);
}

__global__ void k_relu4(float* __restrict__ a, int n4) {
  int i = blockIdx.x * 256 + threadIdx.x;
  if (i >= n4) return;
  float4 v = ((float4*)a)[i];
  v.x = fmaxf(v.x, 0.0f);
  v.y = fmaxf(v.y, 0.0f);
  v.z = fmaxf(v.z, 0.0f);
  v.w = fmaxf(v.w, 0.0f);
  ((float4*)a)[i] = v;
}

// Pack Wcat[k][j] = j<64 ? W2[k][j] : W3[k][j-64]   (128x128)
__global__ void k_pack_w(const float* __restrict__ W2, const float* __restrict__ W3,
                         float* __restrict__ Wc) {
  int t = blockIdx.x * 256 + threadIdx.x;
  if (t >= 128 * 128) return;
  int k = t >> 7;
  int j = t & 127;
  Wc[t] = (j < 64) ? W2[k * 64 + j] : W3[k * 64 + (j - 64)];
}

// ReLU + de-interleave concat result into d_out = [x1 (N x 64) | x2 (N x 64)]
__global__ void k_final(const float* __restrict__ a, float* __restrict__ out, int n) {
  int t = blockIdx.x * 256 + threadIdx.x;
  if (t >= n * 32) return;
  int node = t >> 5;
  int j = (t & 31) << 2;
  float4 v = *(const float4*)(a + (size_t)node * CH + j);
  v.x = fmaxf(v.x, 0.0f);
  v.y = fmaxf(v.y, 0.0f);
  v.z = fmaxf(v.z, 0.0f);
  v.w = fmaxf(v.w, 0.0f);
  float* dst = (j < 64) ? (out + (size_t)node * 64 + j)
                        : (out + (size_t)n * 64 + (size_t)node * 64 + (j - 64));
  *(float4*)dst = v;
}

// ---------------------------------------------------------------------------
extern "C" void kernel_launch(void* const* d_in, const int* in_sizes, int n_in,
                              void* d_out, int out_size, void* d_ws, size_t ws_size,
                              hipStream_t stream) {
  const float* x  = (const float*)d_in[0];
  const int*   ei = (const int*)d_in[1];  // [2, E], int32 per harness convention
  const float* W1 = (const float*)d_in[2];
  const float* b1 = (const float*)d_in[3];
  const float* W2 = (const float*)d_in[4];
  const float* b2 = (const float*)d_in[5];
  const float* W3 = (const float*)d_in[6];
  const float* b3 = (const float*)d_in[7];
  float* out = (float*)d_out;

  const int N = NNODES;
  const int E = NEDGES;
  const int* rows = ei;      // edge_index[0] = source
  const int* cols = ei + E;  // edge_index[1] = target

  // Workspace layout (floats). Total ~51.6 MB.
  float* ws   = (float*)d_ws;
  float* dinv = ws;                          // N (deg -> dinv in place)
  float* wcat = ws + 50176;                  // 128*128
  v16bf* w1hi = (v16bf*)(ws + 66560);        // 1024 frags * 32B = 8192 floats
  v16bf* w1lo = (v16bf*)(ws + 74752);
  v16bf* wchi = (v16bf*)(ws + 82944);
  v16bf* wclo = (v16bf*)(ws + 91136);
  float* bufh = ws + 99328;                  // N*CH : h1, then h23
  float* bufa = bufh + (size_t)N * CH;       // N*CH : agg1/hidden, then agg23

  const int gNode  = (N + 255) / 256;        // 196
  const int gEdge1 = (E + 255) / 256;        // 3125
  const int gVec   = (N * 32 + 255) / 256;   // 6250  (float4 node passes)
  const int gEdgeW = (E * 32 + 255) / 256;   // 100000 (wave per edge)
  const int gGemm  = N / 16;                 // 3125

  // Shared normalization (reference recomputes per conv; identical each time)
  k_deg_init<<<gNode, 256, 0, stream>>>(dinv, N);
  k_deg_count<<<gEdge1, 256, 0, stream>>>(cols, dinv, E);
  k_dinv<<<gNode, 256, 0, stream>>>(dinv, N);

  // Weight prep: Wcat = [W2|W3]; fragment-pack W1 and Wcat (bf16 hi/lo)
  k_pack_w<<<64, 256, 0, stream>>>(W2, W3, wcat);
  k_pack_wfrag<<<4, 256, 0, stream>>>(W1, w1hi, w1lo);
  k_pack_wfrag<<<4, 256, 0, stream>>>(wcat, wchi, wclo);

  // Layer 1: h1 = x @ W1 ; agg = D^-1/2 A D^-1/2 h1 + b1 ; hidden = relu(agg)
  gcn_gemm_bf16x3<<<gGemm, 256, 0, stream>>>(x, w1hi, w1lo, bufh, N);
  k_self_bias<<<gVec, 256, 0, stream>>>(bufh, dinv, b1, b1 + 64, bufa, N);
  k_edge_scatter<<<gEdgeW, 256, 0, stream>>>(rows, cols, dinv, bufh, bufa, E);
  k_relu4<<<gVec, 256, 0, stream>>>(bufa, N * 32);  // bufa = hidden

  // Layers 2+3 fused: h23 = hidden @ [W2|W3]
  gcn_gemm_bf16x3<<<gGemm, 256, 0, stream>>>(bufa, wchi, wclo, bufh, N);
  k_self_bias<<<gVec, 256, 0, stream>>>(bufh, dinv, b2, b3, bufa, N);  // agg23
  k_edge_scatter<<<gEdgeW, 256, 0, stream>>>(rows, cols, dinv, bufh, bufa, E);
  k_final<<<gVec, 256, 0, stream>>>(bufa, out, N);
}
